// PGCU_34497177321831
// MI455X (gfx1250) — matrix operand order repaired and checked by reading, not compiled, not run on hardware
//
#include <hip/hip_runtime.h>

#define BATCH 4
#define CHB   4          // bands C
#define VL    128        // VLEN
#define DD    32         // D = VLEN/C
#define IMG   256
#define EPSLN 1e-5f

typedef __attribute__((ext_vector_type(16))) __bf16 v16bf;
typedef __attribute__((ext_vector_type(8)))  float  v8f;

union ABfrag { v16bf v; unsigned u[8]; };  // 8 VGPRs = 16 bf16

__device__ inline unsigned short f2bf(float f) {
  unsigned u = __float_as_uint(f);
  u += 0x7FFFu + ((u >> 16) & 1u);           // round-to-nearest-even
  return (unsigned short)(u >> 16);
}
__device__ inline unsigned pk2(float lo, float hi) {
  return (unsigned)f2bf(lo) | ((unsigned)f2bf(hi) << 16);
}

// ---------------------------------------------------------------------------
// K0: FmFq = [conv3x3(nearest4x(x)) ; conv3x3(guide)]  -> (B, 8, 256, 256)
// ---------------------------------------------------------------------------
__global__ void k_fmfq(const float* __restrict__ guide, const float* __restrict__ xin,
                       const float* __restrict__ fmw, const float* __restrict__ fmb,
                       const float* __restrict__ fpw, const float* __restrict__ fpb,
                       float* __restrict__ fmfq) {
  int i = blockIdx.x * blockDim.x + threadIdx.x;
  if (i >= BATCH * IMG * IMG) return;
  int b = i >> 16, pix = i & 65535, y = pix >> 8, x = pix & 255;
  const float* xb = xin + b * 4 * 64 * 64;
  const float* gb = guide + b * IMG * IMG;
  float fm[4], fq[4];
  for (int oc = 0; oc < 4; ++oc) { fm[oc] = fmb[oc]; fq[oc] = fpb[oc]; }
  for (int dy = 0; dy < 3; ++dy) {
    int yy = y + dy - 1; if (yy < 0 || yy >= IMG) continue;
    for (int dx = 0; dx < 3; ++dx) {
      int xx = x + dx - 1; if (xx < 0 || xx >= IMG) continue;
      float gv = gb[yy * IMG + xx];
      float uv[4];
      for (int ic = 0; ic < 4; ++ic) uv[ic] = xb[(ic * 64 + (yy >> 2)) * 64 + (xx >> 2)];
      for (int oc = 0; oc < 4; ++oc) {
        float acc = 0.f;
        for (int ic = 0; ic < 4; ++ic) acc += fmw[((oc * 4 + ic) * 3 + dy) * 3 + dx] * uv[ic];
        fm[oc] += acc;
        fq[oc] += fpw[oc * 9 + dy * 3 + dx] * gv;
      }
    }
  }
  for (int oc = 0; oc < 4; ++oc) {
    fmfq[((b * 8 + oc) * IMG + y) * IMG + x]     = fm[oc];
    fmfq[((b * 8 + oc + 4) * IMG + y) * IMG + x] = fq[oc];
  }
}

// ---------------------------------------------------------------------------
// K1: fold fl_w into f_w:  M1[n=cd][k=j] = sum_v fl_w[c,v,d] * f_w[v,j]
// stored pre-swizzled as per-lane WMMA B fragments (bf16), K padded 72->96.
// bias1[cd] = sum_v fl_w[c,v,d] * f_b[v] + fl_b[cd]
// ---------------------------------------------------------------------------
__global__ void k_foldw(const float* __restrict__ flw, const float* __restrict__ flb,
                        const float* __restrict__ fw,  const float* __restrict__ fb,
                        unsigned* __restrict__ m1B, float* __restrict__ bias1) {
  int t = threadIdx.x;
  for (int idx = t; idx < 3 * 8 * 32 * 8; idx += 256) {
    int vp = idx & 7, lane = (idx >> 3) & 31, nt = (idx >> 8) & 7, kk = idx >> 11;
    int n = nt * 16 + (lane & 15);
    int c = n >> 5, d = n & 31;
    int khalf = lane >> 4;
    unsigned out = 0;
    for (int u = 0; u < 2; ++u) {
      int k = kk * 32 + khalf * 16 + vp * 2 + u;
      float s = 0.f;
      if (k < 72)
        for (int v = 0; v < VL; ++v) s += flw[(c * VL + v) * DD + d] * fw[v * 72 + k];
      out |= ((unsigned)f2bf(s)) << (16 * u);
    }
    m1B[idx] = out;
  }
  if (t < VL) {
    int c = t >> 5, d = t & 31;
    float s = flb[t];
    for (int v = 0; v < VL; ++v) s += flw[(c * VL + v) * DD + d] * fb[v];
    bias1[t] = s;
  }
}

// ---------------------------------------------------------------------------
// K2: tiny key-side pipeline (downsample chains, G/V convs, FVF+LN, Vf)
// ---------------------------------------------------------------------------
__device__ inline float at3(const float* p, int H, int W, int c, int y, int x) {
  return (y >= 0 && y < H && x >= 0 && x < W) ? p[(c * H + y) * W + x] : 0.f;
}
// DownSamplingBlock: conv3x3 stride2 pad1 then 2x2 maxpool   H -> H/4, out 4 ch
__device__ void ds_step(const float* in, int IC, int H,
                        const float* w, const float* bias,
                        float* out, int tid, int nt) {
  int OH = H / 4;
  int total = 4 * OH * OH;
  for (int i = tid; i < total; i += nt) {
    int oc = i / (OH * OH), r = i % (OH * OH), oy = r / OH, ox = r % OH;
    float mx = -3.4e38f;
    for (int py = 0; py < 2; ++py)
      for (int px = 0; px < 2; ++px) {
        int cy = 2 * oy + py, cx = 2 * ox + px;
        float acc = bias[oc];
        for (int ic = 0; ic < IC; ++ic)
          for (int dy = 0; dy < 3; ++dy)
            for (int dx = 0; dx < 3; ++dx)
              acc += w[((oc * IC + ic) * 3 + dy) * 3 + dx] *
                     at3(in, H, H, ic, 2 * cy + dy - 1, 2 * cx + dx - 1);
        mx = fmaxf(mx, acc);
      }
    out[i] = mx;
  }
}
__device__ inline float catv(const float* a, const float* bp, int ch, int y, int x) {
  if (y < 0 || y > 3 || x < 0 || x > 3) return 0.f;
  const float* s = (ch < 4) ? a : bp;
  return s[((ch & 3) * 4 + y) * 4 + x];
}

struct K2Args {
  const float *guide, *x;
  const float *gpw0, *gpb0, *gpw1, *gpb1, *gpw2, *gpb2;
  const float *gmw0, *gmb0, *gmw1, *gmb1;
  const float *vpw0, *vpb0, *vpw1, *vpb1, *vpw2, *vpb2;
  const float *vmw0, *vmb0, *vmw1, *vmb1;
  const float *gw, *gb, *vw, *vb;
  const float *glw, *glb, *glg, *glbt;
  float *gpA, *gpB, *GP, *gmA, *GM;
  float *vpA, *vpB, *VP, *vmA, *VM;
  float *GT, *VF;
  unsigned* FVFB;
};

__global__ void k_small(K2Args a) {
  int b = blockIdx.x, t = threadIdx.x;
  const int NT = 256;
  const float* gd = a.guide + b * IMG * IMG;
  const float* xb = a.x + b * 4 * 64 * 64;
  float *gpA = a.gpA + b * 4 * 64 * 64, *gpB = a.gpB + b * 4 * 16 * 16, *GP = a.GP + b * 64;
  float *gmA = a.gmA + b * 4 * 16 * 16, *GM = a.GM + b * 64;
  float *vpA = a.vpA + b * 4 * 64 * 64, *vpB = a.vpB + b * 4 * 16 * 16, *VP = a.VP + b * 64;
  float *vmA = a.vmA + b * 4 * 16 * 16, *VM = a.VM + b * 64;
  float *GT = a.GT + b * VL * 16;

  ds_step(gd, 1, 256, a.gpw0, a.gpb0, gpA, t, NT); __syncthreads();
  ds_step(gpA, 4, 64, a.gpw1, a.gpb1, gpB, t, NT); __syncthreads();
  ds_step(gpB, 4, 16, a.gpw2, a.gpb2, GP, t, NT);  __syncthreads();
  ds_step(xb, 4, 64, a.gmw0, a.gmb0, gmA, t, NT);  __syncthreads();
  ds_step(gmA, 4, 16, a.gmw1, a.gmb1, GM, t, NT);  __syncthreads();
  ds_step(gd, 1, 256, a.vpw0, a.vpb0, vpA, t, NT); __syncthreads();
  ds_step(vpA, 4, 64, a.vpw1, a.vpb1, vpB, t, NT); __syncthreads();
  ds_step(vpB, 4, 16, a.vpw2, a.vpb2, VP, t, NT);  __syncthreads();
  ds_step(xb, 4, 64, a.vmw0, a.vmb0, vmA, t, NT);  __syncthreads();
  ds_step(vmA, 4, 16, a.vmw1, a.vmb1, VM, t, NT);  __syncthreads();

  // G = conv3x3(concat(Gm,Gp)) -> GT[v][k], k = oy*4+ox
  for (int i = t; i < VL * 16; i += NT) {
    int v = i >> 4, k = i & 15, oy = k >> 2, ox = k & 3;
    float acc = a.gb[v];
    for (int c8 = 0; c8 < 8; ++c8)
      for (int dy = 0; dy < 3; ++dy)
        for (int dx = 0; dx < 3; ++dx)
          acc += a.gw[((v * 8 + c8) * 3 + dy) * 3 + dx] * catv(GM, GP, c8, oy + dy - 1, ox + dx - 1);
    GT[i] = acc;
  }
  // V = conv3x3(concat(Vm,Vp)) -> Vf[b][c][k]
  for (int i = t; i < 64; i += NT) {
    int c = i >> 4, k = i & 15, oy = k >> 2, ox = k & 3;
    float acc = a.vb[c];
    for (int c8 = 0; c8 < 8; ++c8)
      for (int dy = 0; dy < 3; ++dy)
        for (int dx = 0; dx < 3; ++dx)
          acc += a.vw[((c * 8 + c8) * 3 + dy) * 3 + dx] * catv(VM, VP, c8, oy + dy - 1, ox + dx - 1);
    a.VF[(b * 4 + c) * 16 + k] = acc;
  }
  __syncthreads();

  // FVF = Gf . gl_w + gl_b, then LayerNorm over d; write swizzled bf16 B-frags
  __shared__ float fvf[CHB][16][DD];
  for (int i = t; i < CHB * 16 * DD; i += NT) {
    int c = i / (16 * DD), r = i % (16 * DD), k = r / DD, d = r % DD;
    float acc = a.glb[c * DD + d];
    for (int v = 0; v < VL; ++v) acc += GT[v * 16 + k] * a.glw[(c * VL + v) * DD + d];
    fvf[c][k][d] = acc;
  }
  __syncthreads();
  if (t < 64) {
    int c = t >> 4, k = t & 15;
    float m = 0.f;
    for (int d = 0; d < DD; ++d) m += fvf[c][k][d];
    m *= (1.f / DD);
    float var = 0.f;
    for (int d = 0; d < DD; ++d) { float df = fvf[c][k][d] - m; var += df * df; }
    var *= (1.f / DD);
    float rs = rsqrtf(var + EPSLN);
    for (int half = 0; half < 2; ++half) {
      int lane = half * 16 + k;             // B layout: lanes 0-15 hold K(d)=0-15
      for (int vp = 0; vp < 8; ++vp) {
        int d0 = half * 16 + vp * 2;
        float n0 = (fvf[c][k][d0] - m) * rs * a.glg[c * DD + d0] + a.glbt[c * DD + d0];
        float n1 = (fvf[c][k][d0 + 1] - m) * rs * a.glg[c * DD + d0 + 1] + a.glbt[c * DD + d0 + 1];
        a.FVFB[((b * 4 + c) * 32 + lane) * 8 + vp] = pk2(n0, n1);
      }
    }
  }
}

// ---------------------------------------------------------------------------
// K3: fused im2col GEMM (WMMA bf16) -> LN -> scores (WMMA) -> softmax -> att
// One wave per 16-query tile; 28 v_wmma_f32_16x16x32_bf16 per tile.
// ---------------------------------------------------------------------------
__global__ void __launch_bounds__(32) k_attn(const float* __restrict__ fmfq,
                                             const unsigned* __restrict__ m1B,
                                             const float* __restrict__ bias1,
                                             const float* __restrict__ flg,
                                             const float* __restrict__ flbt,
                                             const unsigned* __restrict__ fvfB,
                                             const float* __restrict__ vf,
                                             float* __restrict__ att) {
  __shared__ float    halo[8][3][18];     // 8 ch x 3 rows x 18 cols (tile+halo)
  __shared__ unsigned patchA[16][48];     // 16 rows x 96K bf16 (pairs)
  __shared__ float    pvf[16][VL];        // projected queries (f32)
  __shared__ unsigned aln[CHB][16][16];   // LN'd queries, bf16 pairs
  __shared__ float    sc[CHB][16][16];    // scores

  int lane = threadIdx.x;
  int bid = blockIdx.x;
  int b = bid >> 12;                      // 4096 tiles per batch
  int rem = bid & 4095;
  int y = rem >> 4;
  int x0 = (rem & 15) << 4;

  // 1) halo load from FmFq
  for (int i = lane; i < 8 * 3 * 18; i += 32) {
    int ch = i / 54, r = (i % 54) / 18, col = i % 18;
    int gy = y + r - 1, gx = x0 + col - 1;
    float v = 0.f;
    if (gy >= 0 && gy < IMG && gx >= 0 && gx < IMG)
      v = fmfq[((b * 8 + ch) * IMG + gy) * IMG + gx];
    halo[ch][r][col] = v;
  }
  __syncthreads();

  // 2) im2col A-patch in bf16 (K padded 72 -> 96 with zeros)
  for (int i = lane; i < 16 * 48; i += 32) {
    int m = i / 48, dw = i % 48;
    float v0 = 0.f, v1 = 0.f;
    int j0 = dw * 2;
    if (j0 < 72) { int c8 = j0 / 9, tt = j0 % 9; v0 = halo[c8][tt / 3][m + tt % 3]; }
    if (j0 + 1 < 72) { int j1 = j0 + 1; int c8 = j1 / 9, tt = j1 % 9; v1 = halo[c8][tt / 3][m + tt % 3]; }
    patchA[m][dw] = pk2(v0, v1);
  }
  __syncthreads();

  int mrow  = lane & 15;
  int hsel  = (lane >> 4) << 3;   // A-layout: +8 dwords for lanes 16-31
  int rbase = (lane >> 4) << 3;   // D-layout: rows +8 for lanes 16-31

  // 3) per-lane A fragments for the 3 K-chunks (kept in VGPRs)
  ABfrag afr[3];
  for (int kk = 0; kk < 3; ++kk)
    for (int v = 0; v < 8; ++v)
      afr[kk].u[v] = patchA[mrow][kk * 16 + v + hsel];

  // 4) main GEMM: PVF[16q x 128cd] = A[16x96] * M1B[96x128]  (24 WMMAs)
  for (int nt = 0; nt < 8; ++nt) {
    v8f acc = {};
    for (int kk = 0; kk < 3; ++kk) {
      ABfrag bf_;
      const unsigned* bp = m1B + (((kk * 8 + nt) * 32 + lane) << 3);
      uint4 t0 = *(const uint4*)(bp);
      uint4 t1 = *(const uint4*)(bp + 4);
      bf_.u[0] = t0.x; bf_.u[1] = t0.y; bf_.u[2] = t0.z; bf_.u[3] = t0.w;
      bf_.u[4] = t1.x; bf_.u[5] = t1.y; bf_.u[6] = t1.z; bf_.u[7] = t1.w;
      acc = __builtin_amdgcn_wmma_f32_16x16x32_bf16(false, afr[kk].v, false, bf_.v,
                                                    (short)0, acc, false, false);
    }
    float bb = bias1[nt * 16 + mrow];
    for (int i = 0; i < 8; ++i)
      pvf[i + rbase][nt * 16 + mrow] = acc[i] + bb;
  }
  __syncthreads();

  // 5) LayerNorm per (band, query) over D=32, emit bf16 A-rows
  for (int g = lane; g < 64; g += 32) {
    int c = g >> 4, q = g & 15;
    const float* row = &pvf[q][c * DD];
    float m = 0.f;
    for (int d = 0; d < DD; ++d) m += row[d];
    m *= (1.f / DD);
    float var = 0.f;
    for (int d = 0; d < DD; ++d) { float df = row[d] - m; var += df * df; }
    var *= (1.f / DD);
    float rs = rsqrtf(var + EPSLN);
    for (int dw = 0; dw < 16; ++dw) {
      int d0 = dw * 2;
      float n0 = (row[d0] - m) * rs * flg[c * DD + d0] + flbt[c * DD + d0];
      float n1 = (row[d0 + 1] - m) * rs * flg[c * DD + d0 + 1] + flbt[c * DD + d0 + 1];
      aln[c][q][dw] = pk2(n0, n1);
    }
  }
  __syncthreads();

  // 6) scores: per band, (16q x 32d) * (32d x 16k)  (4 WMMAs)
  const float scl = 0.17677669529663689f;  // 1/sqrt(32)
  for (int c = 0; c < CHB; ++c) {
    ABfrag a2, b2;
    for (int v = 0; v < 8; ++v)
      a2.u[v] = aln[c][mrow][v + hsel];
    const unsigned* bp = fvfB + (((b * 4 + c) * 32 + lane) << 3);
    uint4 t0 = *(const uint4*)(bp);
    uint4 t1 = *(const uint4*)(bp + 4);
    b2.u[0] = t0.x; b2.u[1] = t0.y; b2.u[2] = t0.z; b2.u[3] = t0.w;
    b2.u[4] = t1.x; b2.u[5] = t1.y; b2.u[6] = t1.z; b2.u[7] = t1.w;
    v8f acc = {};
    acc = __builtin_amdgcn_wmma_f32_16x16x32_bf16(false, a2.v, false, b2.v,
                                                  (short)0, acc, false, false);
    for (int i = 0; i < 8; ++i)
      sc[c][i + rbase][mrow] = acc[i] * scl;
  }
  __syncthreads();

  // 7) softmax over 16 keys + weighted sum with Vf
  for (int g = lane; g < 64; g += 32) {
    int c = g >> 4, q = g & 15;
    float mx = -3.4e38f;
    for (int k = 0; k < 16; ++k) mx = fmaxf(mx, sc[c][q][k]);
    float s = 0.f, o = 0.f;
    for (int k = 0; k < 16; ++k) {
      float e = __expf(sc[c][q][k] - mx);
      s += e;
      o += e * vf[(b * 4 + c) * 16 + k];
    }
    att[((b * 4 + c) * IMG + y) * IMG + x0 + q] = o / s;
  }
}

// ---------------------------------------------------------------------------
// K4: final fa conv3x3 -> d_out
// ---------------------------------------------------------------------------
__global__ void k_faconv(const float* __restrict__ att, const float* __restrict__ faw,
                         const float* __restrict__ fab, float* __restrict__ out) {
  int i = blockIdx.x * blockDim.x + threadIdx.x;
  if (i >= BATCH * CHB * IMG * IMG) return;
  int x = i & 255, y = (i >> 8) & 255, oc = (i >> 16) & 3, b = i >> 18;
  float acc = fab[oc];
  for (int ic = 0; ic < 4; ++ic)
    for (int dy = 0; dy < 3; ++dy) {
      int yy = y + dy - 1; if (yy < 0 || yy >= IMG) continue;
      for (int dx = 0; dx < 3; ++dx) {
        int xx = x + dx - 1; if (xx < 0 || xx >= IMG) continue;
        acc += faw[((oc * 4 + ic) * 3 + dy) * 3 + dx] * att[((b * 4 + ic) * IMG + yy) * IMG + xx];
      }
    }
  out[i] = acc;
}

// ---------------------------------------------------------------------------
// Host launcher
// ---------------------------------------------------------------------------
enum {
  P_F_B, P_F_W, P_FA_B, P_FA_W, P_FL_B, P_FL_BETA, P_FL_G, P_FL_W,
  P_FM_B, P_FM_W, P_FP_B, P_FP_W, P_G_B, P_G_W,
  P_GL_B, P_GL_BETA, P_GL_G, P_GL_W,
  P_GM0_B, P_GM0_W, P_GM1_B, P_GM1_W,
  P_GP0_B, P_GP0_W, P_GP1_B, P_GP1_W, P_GP2_B, P_GP2_W,
  P_V_B, P_V_W,
  P_VM0_B, P_VM0_W, P_VM1_B, P_VM1_W,
  P_VP0_B, P_VP0_W, P_VP1_B, P_VP1_W, P_VP2_B, P_VP2_W,
  P_COUNT
};

extern "C" void kernel_launch(void* const* d_in, const int* in_sizes, int n_in,
                              void* d_out, int out_size, void* d_ws, size_t ws_size,
                              hipStream_t stream) {
  const float* guide = (const float*)d_in[0];
  const float* x     = (const float*)d_in[1];

  // params pytree flattening: sorted-dict (jax tree_leaves) vs insertion order
  static const int sortedIdx[P_COUNT] = {
    2, 3, 4, 5, 6, 7, 8, 9, 10, 11, 12, 13, 14, 15, 16, 17, 18, 19,
    20, 21, 22, 23, 24, 25, 26, 27, 28, 29, 30, 31, 32, 33, 34, 35,
    36, 37, 38, 39, 40, 41
  };
  static const int insIdx[P_COUNT] = {
    7, 6, 13, 12, 35, 37, 36, 34, 5, 4, 3, 2, 9, 8,
    39, 41, 40, 38, 21, 20, 23, 22, 15, 14, 17, 16, 19, 18,
    11, 10, 31, 30, 33, 32, 25, 24, 27, 26, 29, 28
  };
  const int* pm = (in_sizes[2] == 128) ? sortedIdx : insIdx;
  const float* P[P_COUNT];
  for (int i = 0; i < P_COUNT; ++i) P[i] = (const float*)d_in[pm[i]];

  // workspace layout (floats)
  float* wsf = (float*)d_ws;
  size_t off = 0;
  float*    FMFQ = wsf + off; off += (size_t)BATCH * 8 * IMG * IMG;   // 2,097,152
  float*    ATT  = wsf + off; off += (size_t)BATCH * 4 * IMG * IMG;   // 1,048,576
  unsigned* M1B  = (unsigned*)(wsf + off); off += 3 * 8 * 32 * 8;     // 6144
  float*    B1   = wsf + off; off += VL;                              // 128
  unsigned* FVFB = (unsigned*)(wsf + off); off += BATCH * 4 * 32 * 8; // 4096
  float*    VF   = wsf + off; off += BATCH * 4 * 16;                  // 256
  float* GPA = wsf + off; off += (size_t)BATCH * 4 * 64 * 64;
  float* GPB = wsf + off; off += (size_t)BATCH * 4 * 16 * 16;
  float* GP  = wsf + off; off += BATCH * 64;
  float* GMA = wsf + off; off += (size_t)BATCH * 4 * 16 * 16;
  float* GM  = wsf + off; off += BATCH * 64;
  float* VPA = wsf + off; off += (size_t)BATCH * 4 * 64 * 64;
  float* VPB = wsf + off; off += (size_t)BATCH * 4 * 16 * 16;
  float* VP  = wsf + off; off += BATCH * 64;
  float* VMA = wsf + off; off += (size_t)BATCH * 4 * 16 * 16;
  float* VM  = wsf + off; off += BATCH * 64;
  float* GT  = wsf + off; off += (size_t)BATCH * VL * 16;

  // K0: FmFq
  k_fmfq<<<(BATCH * IMG * IMG + 255) / 256, 256, 0, stream>>>(
      guide, x, P[P_FM_W], P[P_FM_B], P[P_FP_W], P[P_FP_B], FMFQ);

  // K1: fold fl_w into f_w -> swizzled bf16 B + bias
  k_foldw<<<1, 256, 0, stream>>>(P[P_FL_W], P[P_FL_B], P[P_F_W], P[P_F_B], M1B, B1);

  // K2: key-side pipeline
  K2Args a;
  a.guide = guide; a.x = x;
  a.gpw0 = P[P_GP0_W]; a.gpb0 = P[P_GP0_B];
  a.gpw1 = P[P_GP1_W]; a.gpb1 = P[P_GP1_B];
  a.gpw2 = P[P_GP2_W]; a.gpb2 = P[P_GP2_B];
  a.gmw0 = P[P_GM0_W]; a.gmb0 = P[P_GM0_B];
  a.gmw1 = P[P_GM1_W]; a.gmb1 = P[P_GM1_B];
  a.vpw0 = P[P_VP0_W]; a.vpb0 = P[P_VP0_B];
  a.vpw1 = P[P_VP1_W]; a.vpb1 = P[P_VP1_B];
  a.vpw2 = P[P_VP2_W]; a.vpb2 = P[P_VP2_B];
  a.vmw0 = P[P_VM0_W]; a.vmb0 = P[P_VM0_B];
  a.vmw1 = P[P_VM1_W]; a.vmb1 = P[P_VM1_B];
  a.gw = P[P_G_W]; a.gb = P[P_G_B];
  a.vw = P[P_V_W]; a.vb = P[P_V_B];
  a.glw = P[P_GL_W]; a.glb = P[P_GL_B]; a.glg = P[P_GL_G]; a.glbt = P[P_GL_BETA];
  a.gpA = GPA; a.gpB = GPB; a.GP = GP; a.gmA = GMA; a.GM = GM;
  a.vpA = VPA; a.vpB = VPB; a.VP = VP; a.vmA = VMA; a.VM = VM;
  a.GT = GT; a.VF = VF; a.FVFB = FVFB;
  k_small<<<BATCH, 256, 0, stream>>>(a);

  // K3: fused WMMA attention (16384 tiles, 1 wave each)
  k_attn<<<BATCH * 16 * IMG, 32, 0, stream>>>(
      FMFQ, M1B, B1, P[P_FL_G], P[P_FL_BETA], FVFB, VF, ATT);

  // K4: final conv -> d_out
  k_faconv<<<(BATCH * CHB * IMG * IMG + 255) / 256, 256, 0, stream>>>(
      ATT, P[P_FA_W], P[P_FA_B], (float*)d_out);
}
